// JointNetwork_39874476376809
// MI455X (gfx1250) — matrix operand order repaired and checked by reading, not compiled, not run on hardware
//
#include <hip/hip_runtime.h>
#include <hip/hip_bf16.h>
#include <math.h>

// ---------------------------------------------------------------------------
// RNN-T joint network on MI455X (gfx1250, wave32, WMMA + TDM).
//   h_enc = x_enc @ w_l + b_l            (1600 x 512)
//   h_prd = x_prd @ w_p + b_p            ( 400 x 512)
//   h     = tanh(h_enc[bt] + h_prd[bu])  (80000 x 512, f16)
//   out   = h @ w_h + b_h                (80000 x 1024, f32)
// GEMMs: v_wmma_f32_16x16x32_f16, A tile DMA'd to LDS by the Tensor Data
// Mover (tensor_load_to_lds, TENSORcnt), double-buffered.
// ---------------------------------------------------------------------------

typedef __attribute__((ext_vector_type(16))) _Float16 v16h;
typedef __attribute__((ext_vector_type(8)))  _Float16 v8h;
typedef __attribute__((ext_vector_type(8)))  float    v8f;
typedef __attribute__((ext_vector_type(4)))  float    v4f;
typedef __attribute__((ext_vector_type(4)))  unsigned int v4u;
typedef __attribute__((ext_vector_type(8)))  int      v8i;
typedef __attribute__((ext_vector_type(4)))  int      v4i;

union V16 { v16h v; v8h h[2]; };

#define BM 128
#define BN 128
#define KDIM 512
#define KSTEPS 16
#define LDA_PAD 40   // 32 halfs + 8 pad -> 80B row stride (TDM pad_amount)

// ---- elementwise f32 -> f16 ------------------------------------------------
__global__ void cvt_f32_to_f16(const float* __restrict__ src,
                               _Float16* __restrict__ dst, int n) {
  int i = blockIdx.x * blockDim.x + threadIdx.x;
  int stride = gridDim.x * blockDim.x;
  for (; i < n; i += stride) dst[i] = (_Float16)src[i];
}

// ---- swizzle W (K=512 x N, f32 row-major) into WMMA-B fragment order ------
// dst layout: [ks][ntile][lane][j] (f16), B 32x16 f16 tile:
//   col = ntile*16 + (lane & 15)
//   k   = ks*32 + (j < 8 ? 0 : 16) + (lane < 16 ? 0 : 8) + (j & 7)
__global__ void swizzle_w(const float* __restrict__ w,
                          _Float16* __restrict__ dst, int N) {
  int total = KDIM * N;
  int ntiles = N >> 4;
  int i = blockIdx.x * blockDim.x + threadIdx.x;
  int stride = gridDim.x * blockDim.x;
  for (; i < total; i += stride) {
    int j    = i & 15;
    int lane = (i >> 4) & 31;
    int nt   = (i >> 9) % ntiles;
    int ks   = i / (512 * ntiles);
    int k = ks * 32 + ((j >> 3) << 4) + ((lane >> 4) << 3) + (j & 7);
    int n = nt * 16 + (lane & 15);
    dst[i] = (_Float16)w[k * N + n];
  }
}

// ---- h = tanh(h_enc[b*T+t] + h_prd[b*U+u]) -> f16, 8 elems/thread --------
__global__ void combine_tanh(const float* __restrict__ hE,
                             const float* __restrict__ hP,
                             _Float16* __restrict__ h16) {
  const int total = (80000 * 512) / 8;
  int g = blockIdx.x * blockDim.x + threadIdx.x;
  int stride = gridDim.x * blockDim.x;
  for (; g < total; g += stride) {
    int m  = g >> 6;          // row (b*200+t)*50+u
    int kb = (g & 63) << 3;   // k base (8 halfs)
    int rowE = m / 50;
    int rowP = (m / 10000) * 50 + (m % 50);
    const v4f* pe = (const v4f*)(hE + rowE * 512 + kb);
    const v4f* pp = (const v4f*)(hP + rowP * 512 + kb);
    v4f e0 = pe[0], e1 = pe[1], p0 = pp[0], p1 = pp[1];
    v8h r;
#pragma unroll
    for (int j = 0; j < 4; ++j) r[j] = (_Float16)tanhf(e0[j] + p0[j]);
#pragma unroll
    for (int j = 0; j < 4; ++j) r[4 + j] = (_Float16)tanhf(e1[j] + p1[j]);
    *(v8h*)(h16 + ((size_t)g << 3)) = r;
  }
}

// ---- TDM: DMA a BM x 32-half A tile (row stride 512 halfs) into LDS ------
// D# per ISA ch.8: 2D tile 32 x 128 elements of 2B, pad 4 DWORDs (16B) after
// every 16 DWORDs (64B) -> LDS row stride 80B. tensor_dim1 = rows_valid gives
// hardware zero-fill past the M edge.
__device__ inline void tdm_load_a_tile(const _Float16* gsrc,
                                       unsigned lds_off, int rows_valid) {
  unsigned long long ga = (unsigned long long)(uintptr_t)gsrc;
  v4u g0;
  g0[0] = 1u;                                        // count=1 (valid), user
  g0[1] = lds_off;                                   // lds_addr (bytes)
  g0[2] = (unsigned)(ga & 0xffffffffu);              // global_addr[31:0]
  g0[3] = (unsigned)((ga >> 32) & 0x01ffffffu)       // global_addr[56:32]
          | (2u << 30);                              // type = 2 (image)
  unsigned rv = (unsigned)rows_valid;
  v8i g1;
  g1[0] = (int)((1u << 16)      // data_size = 2 bytes
              | (1u << 20)      // pad_enable
              | (3u << 22)      // pad_interval: 16 DWORDs (64B)
              | (3u << 25));    // pad_amount:   4 DWORDs (16B)
  g1[1] = (int)(512u << 16);                         // tensor_dim0 lo16 = 512
  g1[2] = (int)((rv & 0xffffu) << 16);               // dim0 hi=0 | dim1 lo16
  g1[3] = (int)((rv >> 16) | (32u << 16));           // dim1 hi16 | tile_dim0=32
  g1[4] = 128;                                       // tile_dim1=128, dim2=0
  g1[5] = 512;                                       // tensor_dim0_stride lo32
  g1[6] = 0;                                         // stride hi | dim1_stride
  g1[7] = 0;
  v4i z4 = {0, 0, 0, 0};
  v8i z8 = {0, 0, 0, 0, 0, 0, 0, 0};
  // 6-arg form (clang-23 / therock-10.0 headers)
  __builtin_amdgcn_tensor_load_to_lds(g0, g1, z4, z4, z8, 0);
}

// ---- tiled WMMA GEMM: C[M,N] = A[M,512] @ Wswz + bias ---------------------
// 256 threads = 8 wave32s; block tile 128x128; wave tile 32x64 (2x4 WMMA).
// A tile: TDM double-buffered LDS. B frags: pre-swizzled global (L2).
__global__ __launch_bounds__(256) void gemm_wmma_f16(
    const _Float16* __restrict__ A,     // M x 512, row-major f16
    const _Float16* __restrict__ Wswz,  // [16][N/16][32][16] f16
    const float* __restrict__ bias,     // N
    float* __restrict__ C,              // M x N f32
    int M, int N) {
  __shared__ _Float16 As[2][BM * LDA_PAD];

  const int tid    = threadIdx.x;
  const int lane   = tid & 31;
  const int wid    = tid >> 5;
  const int wm     = wid & 3;   // 4 wave-rows of 32
  const int wn     = wid >> 2;  // 2 wave-cols of 64
  const int mB     = blockIdx.x * BM;
  const int nB     = blockIdx.y * BN;
  const int laneLo = lane & 15;
  const int laneHi = lane >> 4;
  const int ntilesN = N >> 4;
  const int rowsValid = M - mB;                  // hw zero-fill past edge
  const unsigned lds0 = (unsigned)(uintptr_t)(&As[0][0]);
  const unsigned lds1 = (unsigned)(uintptr_t)(&As[1][0]);
  const _Float16* Ablk = A + (size_t)mB * KDIM;

  v8f acc[2][4];
#pragma unroll
  for (int i = 0; i < 2; ++i)
#pragma unroll
    for (int j = 0; j < 4; ++j) acc[i][j] = {};

  // prologue: kick off the first A tile DMA (one TDM op per workgroup)
  if (wid == 0) tdm_load_a_tile(Ablk, lds0, rowsValid);

  for (int ks = 0; ks < KSTEPS; ++ks) {
    const int cur = ks & 1;
    if (wid == 0) {
      if (ks + 1 < KSTEPS) {
        // prefetch next tile into the other buffer, then wait for current
        tdm_load_a_tile(Ablk + (ks + 1) * 32, cur ? lds0 : lds1, rowsValid);
        __builtin_amdgcn_s_wait_tensorcnt(1);
      } else {
        __builtin_amdgcn_s_wait_tensorcnt(0);
      }
    }
    __syncthreads();   // release all waves once current tile landed

    // ---- A fragments (16-bit A 16x32 layout, ISA 7.12.2) ----------------
    V16 afrag[2];
#pragma unroll
    for (int mt = 0; mt < 2; ++mt) {
      int r = wm * 32 + mt * 16 + laneLo;
      const _Float16* p = &As[cur][r * LDA_PAD + laneHi * 8];
      afrag[mt].h[0] = *(const v8h*)(p);
      afrag[mt].h[1] = *(const v8h*)(p + 16);
    }

    // ---- B fragments from pre-swizzled weights (contiguous 32B/lane) ----
    V16 bfrag[4];
#pragma unroll
    for (int nt = 0; nt < 4; ++nt) {
      int ntg = (nB >> 4) + wn * 4 + nt;
      const _Float16* p =
          Wswz + (((size_t)ks * ntilesN + ntg) * 32 + lane) * 16;
      bfrag[nt].h[0] = *(const v8h*)(p);
      bfrag[nt].h[1] = *(const v8h*)(p + 8);
    }

    // ---- 8 WMMAs per wave per K-step ------------------------------------
#pragma unroll
    for (int mt = 0; mt < 2; ++mt)
#pragma unroll
      for (int nt = 0; nt < 4; ++nt)
        acc[mt][nt] = __builtin_amdgcn_wmma_f32_16x16x32_f16(
            false, afrag[mt].v, false, bfrag[nt].v, (short)0, acc[mt][nt],
            false, false);

    // all waves done reading As[cur] before TDM retargets it next iter
    __syncthreads();
  }

  // ---- epilogue: bias add + f32 store (C/D layout: row = r + 8*laneHi) ---
#pragma unroll
  for (int mt = 0; mt < 2; ++mt) {
#pragma unroll
    for (int nt = 0; nt < 4; ++nt) {
      int col = nB + wn * 64 + nt * 16 + laneLo;
      float bv = bias[col];
#pragma unroll
      for (int r = 0; r < 8; ++r) {
        int row = mB + wm * 32 + mt * 16 + laneHi * 8 + r;
        if (row < M) C[(size_t)row * N + col] = acc[mt][nt][r] + bv;
      }
    }
  }
}

// ---------------------------------------------------------------------------
extern "C" void kernel_launch(void* const* d_in, const int* in_sizes, int n_in,
                              void* d_out, int out_size, void* d_ws,
                              size_t ws_size, hipStream_t stream) {
  const float* x_enc = (const float*)d_in[0];  // (8,200,1,512)
  const float* x_prd = (const float*)d_in[1];  // (8,1,50,512)
  const float* w_l   = (const float*)d_in[2];  // (512,512)
  const float* b_l   = (const float*)d_in[3];  // (512)
  const float* w_p   = (const float*)d_in[4];  // (512,512)
  const float* b_p   = (const float*)d_in[5];  // (512)
  const float* w_h   = (const float*)d_in[6];  // (512,1024)
  const float* b_h   = (const float*)d_in[7];  // (1024)
  float* out = (float*)d_out;                  // (8,200,50,1024)

  char* ws = (char*)d_ws;
  size_t o = 0;
  auto take = [&](size_t bytes) -> char* {
    char* p = ws + o;
    o = (o + bytes + 255) & ~(size_t)255;
    return p;
  };
  _Float16* xe16 = (_Float16*)take((size_t)819200 * 2);
  _Float16* xp16 = (_Float16*)take((size_t)204800 * 2);
  _Float16* wl_s = (_Float16*)take((size_t)262144 * 2);
  _Float16* wp_s = (_Float16*)take((size_t)262144 * 2);
  _Float16* wh_s = (_Float16*)take((size_t)524288 * 2);
  float*    hE   = (float*)take((size_t)819200 * 4);
  float*    hP   = (float*)take((size_t)204800 * 4);
  _Float16* h16  = (_Float16*)take((size_t)40960000 * 2);

  // precision prep: f16 A matrices + fragment-swizzled f16 weights
  cvt_f32_to_f16<<<2048, 256, 0, stream>>>(x_enc, xe16, 819200);
  cvt_f32_to_f16<<<1024, 256, 0, stream>>>(x_prd, xp16, 204800);
  swizzle_w<<<1024, 256, 0, stream>>>(w_l, wl_s, 512);
  swizzle_w<<<1024, 256, 0, stream>>>(w_p, wp_s, 512);
  swizzle_w<<<2048, 256, 0, stream>>>(w_h, wh_s, 1024);

  // projections (WMMA f16 -> f32)
  gemm_wmma_f16<<<dim3(13, 4), 256, 0, stream>>>(xe16, wl_s, b_l, hE, 1600, 512);
  gemm_wmma_f16<<<dim3(4, 4), 256, 0, stream>>>(xp16, wp_s, b_p, hP, 400, 512);

  // broadcast add + tanh, materialize h once in f16 (L2-resident, 82 MB)
  combine_tanh<<<4096, 256, 0, stream>>>(hE, hP, h16);

  // joint GEMM: 80000 x 1024 x 512, HBM-write bound (~328 MB out)
  gemm_wmma_f16<<<dim3(625, 8), 256, 0, stream>>>(h16, wh_s, b_h, out, 80000, 1024);
}
